// DSS_31817117729292
// MI455X (gfx1250) — matrix operand order repaired and checked by reading, not compiled, not run on hardware
//
#include <hip/hip_runtime.h>
#include <hip/hip_bf16.h>

// ---------------------------------------------------------------------------
// DSS forward for MI455X (gfx1250, wave32, WMMA).
//   B=16, H=512, L=2048, N=64, C=1
// Pipeline:
//   1) dss_params : (h,n) -> lambda, c   (exact reference math incl. pos/P_max)
//   2) pack_w     : out_weight f32 -> bf16 WMMA A-fragment layout
//   3) dss_scan   : state-space scan == FFT causal conv, + D*u skip + erf-GELU,
//                   writes bf16 WMMA B-fragment layout
//   4) dss_gemm   : 512x512x32768 GEMM, 2x2 register-blocked
//                   (4x v_wmma_f32_16x16x32_bf16 per K-step, 2 b128 loads/WMMA)
// ---------------------------------------------------------------------------

typedef __bf16 bf16_t;
typedef __attribute__((ext_vector_type(16))) __bf16 v16bf;
typedef __attribute__((ext_vector_type(8)))  float  v8f;

namespace {
constexpr int kB = 16;
constexpr int kH = 512;
constexpr int kL = 2048;
constexpr int kN = 64;
constexpr int kLT = kL / 16;   // 128 l-tiles per batch
constexpr int kKT = kH / 32;   // 16 k-tiles over H (GEMM K)
}

// --- Kernel 1: per-(h,n) DSS mode parameters --------------------------------
// params[(h*N+n)*4] = { lambda_re, lambda_im, c_re, c_im }
__global__ void dss_params_kernel(const float* __restrict__ log_dt,
                                  const float* __restrict__ Lambda,
                                  const float* __restrict__ W,
                                  float* __restrict__ params) {
  int t = blockIdx.x * blockDim.x + threadIdx.x;    // [0, H*N)
  int h = t / kN;
  int n = t % kN;

  float dtr = expf(log_dt[h * 2 + 0]);
  float dti = expf(log_dt[h * 2 + 1]);
  float Lr = Lambda[n * 2 + 0];
  float Li = Lambda[n * 2 + 1];

  // dt_Lambda (sep_dt_re_im)
  float ar = dtr * Lr;
  float ai = dti * Li;
  float pos = (Lr > 0.0f) ? 1.0f : 0.0f;

  // lambda = exp(dt_Lambda)
  float er = expf(ar);
  float lr = er * cosf(ai);
  float li = er * sinf(ai);

  // dt_Lambda_neg = dt_Lambda * (1 - 2*pos)
  float nf  = 1.0f - 2.0f * pos;
  float anr = ar * nf;
  float ani = ai * nf;

  // num = exp(dt_Lambda_neg) - 1
  float enr  = expf(anr);
  float numr = enr * cosf(ani) - 1.0f;
  float numi = enr * sinf(ani);

  // den = exp(dt_Lambda_neg * L) - 1 ; x = den * Lambda
  float Lf   = (float)kL;
  float eLr  = expf(anr * Lf);
  float denr = eLr * cosf(ani * Lf) - 1.0f;
  float deni = eLr * sinf(ani * Lf);
  float xr = denr * Lr - deni * Li;
  float xi = denr * Li + deni * Lr;

  // recip = conj(x) / (|x|^2 + EPS)
  float mag = xr * xr + xi * xi + 1e-7f;
  float rr  =  xr / mag;
  float ri  = -xi / mag;

  // Wk = W * num * recip
  float Wr = W[(h * kN + n) * 2 + 0];
  float Wi = W[(h * kN + n) * 2 + 1];
  float tr = Wr * numr - Wi * numi;
  float ti = Wr * numi + Wi * numr;
  float cr0 = tr * rr - ti * ri;
  float ci0 = tr * ri + ti * rr;

  // fold exp(-P_max) into c, P_max = dt_Lambda * pos * (L-1)
  float pm  = pos * (Lf - 1.0f);
  float smg = expf(-ar * pm);
  float scr =  smg * cosf(ai * pm);
  float sci = -smg * sinf(ai * pm);
  float cr = cr0 * scr - ci0 * sci;
  float ci = cr0 * sci + ci0 * scr;

  float* o = params + (size_t)t * 4;
  o[0] = lr; o[1] = li; o[2] = cr; o[3] = ci;
}

// --- Kernel 2: out_weight -> bf16 WMMA A-fragment layout --------------------
// Wfrag[((mt*kKT + kt)*32 + lane)*16 + j] = bf16(W[m, k])
//   m = mt*16 + (lane&15)
//   k = kt*32 + (lane>>4)*16 + j       (contiguous k order, same as Bfrag)
__global__ void pack_w_kernel(const float* __restrict__ Wout,
                              bf16_t* __restrict__ Wfrag) {
  int t = blockIdx.x * blockDim.x + threadIdx.x;    // [0, H*H)
  int j    =  t        & 15;
  int lane = (t >> 4)  & 31;
  int kt   = (t >> 9)  & (kKT - 1);
  int mt   =  t >> 13;
  int m = mt * 16 + (lane & 15);
  int k = kt * 32 + ((lane >> 4) * 16) + j;
  Wfrag[t] = (bf16_t)Wout[(size_t)m * kH + k];
}

// --- Kernel 3: state-space scan (== FFT causal conv) + skip + GELU ----------
// One wave32 per (b,h); lane owns modes n=lane and n=lane+32.
// Writes gelu(conv + D*u) as bf16 into WMMA B-fragment layout:
//   Bfrag[(((b*kLT + lt)*kKT + kt)*32 + laneB)*16 + j]
//   kt = h>>5, laneB = (l&15) + 16*((h>>4)&1), j = h&15, lt = l>>4
__global__ void dss_scan_kernel(const float* __restrict__ u,
                                const float* __restrict__ params,
                                const float* __restrict__ D,
                                bf16_t* __restrict__ Bfrag) {
  int wave = (blockIdx.x * blockDim.x + threadIdx.x) >> 5;  // [0, B*H)
  int lane = threadIdx.x & 31;
  int b = wave / kH;
  int h = wave % kH;

  const float* p0 = params + ((size_t)h * kN + lane) * 4;
  const float* p1 = params + ((size_t)h * kN + lane + 32) * 4;
  float l0r = p0[0], l0i = p0[1], c0r = p0[2], c0i = p0[3];
  float l1r = p1[0], l1i = p1[1], c1r = p1[2], c1i = p1[3];
  float s0r = 0.f, s0i = 0.f, s1r = 0.f, s1i = 0.f;
  float Dh = D[h];

  const float* up = u + ((size_t)b * kH + h) * kL;
  int kt    = h >> 5;
  int laneB = ((h >> 4) & 1) * 16;
  int j     = h & 15;

  for (int base = 0; base < kL; base += 32) {
    float uv = up[base + lane];          // coalesced 32-wide chunk
    float ymine = 0.0f;
    for (int i = 0; i < 32; ++i) {
      float ub = __shfl(uv, i, 32);      // broadcast u[base+i]
      // complex state updates: s = lambda*s + u
      float t0r = l0r * s0r - l0i * s0i + ub;
      float t0i = l0r * s0i + l0i * s0r;
      s0r = t0r; s0i = t0i;
      float t1r = l1r * s1r - l1i * s1i + ub;
      float t1i = l1r * s1i + l1i * s1r;
      s1r = t1r; s1i = t1i;
      // Re(c * s), both modes
      float contrib = c0r * s0r - c0i * s0i + c1r * s1r - c1i * s1i;
      // wave32 xor-reduction over 64 modes (2 per lane)
      contrib += __shfl_xor(contrib, 1, 32);
      contrib += __shfl_xor(contrib, 2, 32);
      contrib += __shfl_xor(contrib, 4, 32);
      contrib += __shfl_xor(contrib, 8, 32);
      contrib += __shfl_xor(contrib, 16, 32);
      float y = contrib + Dh * ub;                        // skip connection
      float g = 0.5f * y * (1.0f + erff(y * 0.70710678118654752f)); // exact GELU
      if (lane == i) ymine = g;          // register transpose: lane i keeps step i
    }
    int l    = base + lane;
    int lt   = l >> 4;
    int nloc = l & 15;
    size_t off = ((((size_t)b * kLT + lt) * kKT + kt) * 32 + (nloc + laneB)) * 16 + j;
    Bfrag[off] = (bf16_t)ymine;
  }
}

// --- Kernel 4: GEMM  out[b,v,l] = sum_u W[v,u]*yact[b,u,l] + bias[v] --------
// 2x2 register-blocked: one wave computes a 32(M) x 32(N) output block as
// four 16x16 WMMA tiles. Per K-step: 2 A-frags + 2 B-frags (8 b128 loads),
// 4 independent v_wmma_f32_16x16x32_bf16 (also hides WMMA->WMMA hazards).
__global__ void dss_gemm_kernel(const bf16_t* __restrict__ Wfrag,
                                const bf16_t* __restrict__ Bfrag,
                                const float* __restrict__ bias,
                                float* __restrict__ out) {
  int lane = threadIdx.x & 31;
  int w    = threadIdx.x >> 5;
  int np   = blockIdx.x * 8 + w;        // [0, 1024) n-tile pairs (b,l)
  int mp   = blockIdx.y;                // [0, 16)   m-tile pairs (v)

  const bf16_t* a0 = Wfrag + ((size_t)(2 * mp) * kKT) * 512 + lane * 16;
  const bf16_t* a1 = a0 + (size_t)kKT * 512;
  const bf16_t* b0 = Bfrag + ((size_t)(2 * np) * kKT) * 512 + lane * 16;
  const bf16_t* b1 = b0 + (size_t)kKT * 512;

  v8f acc00 = {}, acc01 = {}, acc10 = {}, acc11 = {};
#pragma unroll
  for (int kt = 0; kt < kKT; ++kt) {
    v16bf A0 = *(const v16bf*)(a0 + (size_t)kt * 512);
    v16bf A1 = *(const v16bf*)(a1 + (size_t)kt * 512);
    v16bf B0 = *(const v16bf*)(b0 + (size_t)kt * 512);
    v16bf B1 = *(const v16bf*)(b1 + (size_t)kt * 512);
    if (kt + 1 < kKT) {
      __builtin_prefetch((const void*)(a0 + (size_t)(kt + 1) * 512), 0, 1);
      __builtin_prefetch((const void*)(a1 + (size_t)(kt + 1) * 512), 0, 1);
      __builtin_prefetch((const void*)(b0 + (size_t)(kt + 1) * 512), 0, 1);
      __builtin_prefetch((const void*)(b1 + (size_t)(kt + 1) * 512), 0, 1);
    }
    acc00 = __builtin_amdgcn_wmma_f32_16x16x32_bf16(false, A0, false, B0,
                                                    (short)0, acc00, false, false);
    acc01 = __builtin_amdgcn_wmma_f32_16x16x32_bf16(false, A0, false, B1,
                                                    (short)0, acc01, false, false);
    acc10 = __builtin_amdgcn_wmma_f32_16x16x32_bf16(false, A1, false, B0,
                                                    (short)0, acc10, false, false);
    acc11 = __builtin_amdgcn_wmma_f32_16x16x32_bf16(false, A1, false, B1,
                                                    (short)0, acc11, false, false);
  }

  // epilogue: C/D f32 layout -> lanes 0-15: M=r, lanes 16-31: M=r+8; N=lane&15
  v8f accs[2][2] = {{acc00, acc01}, {acc10, acc11}};
#pragma unroll
  for (int mi = 0; mi < 2; ++mi) {
    int mrow = (2 * mp + mi) * 16 + (lane >> 4) * 8;
#pragma unroll
    for (int ni = 0; ni < 2; ++ni) {
      int nt = 2 * np + ni;
      int b  = nt >> 7;                 // nt / kLT
      int lt = nt & (kLT - 1);
      int l  = lt * 16 + (lane & 15);
#pragma unroll
      for (int r = 0; r < 8; ++r) {
        int v = mrow + r;
        out[(((size_t)b * kH) + v) * kL + l] = accs[mi][ni][r] + bias[v];
      }
    }
  }
}

// ---------------------------------------------------------------------------
extern "C" void kernel_launch(void* const* d_in, const int* in_sizes, int n_in,
                              void* d_out, int out_size, void* d_ws, size_t ws_size,
                              hipStream_t stream) {
  const float* u          = (const float*)d_in[0];  // (B,H,L)
  const float* log_dt     = (const float*)d_in[1];  // (H,2)
  const float* Lambda     = (const float*)d_in[2];  // (N,2)
  const float* W          = (const float*)d_in[3];  // (1,H,N,2)
  const float* D          = (const float*)d_in[4];  // (1,H)
  const float* out_weight = (const float*)d_in[5];  // (H,H)
  const float* out_bias   = (const float*)d_in[6];  // (H,1)
  float* out = (float*)d_out;                       // (B,H,L) f32

  char* ws = (char*)d_ws;
  float*  params = (float*)ws;                          // H*N*4 f32   = 512 KB
  bf16_t* Wfrag  = (bf16_t*)(ws + (512u << 10));        // H*H bf16    = 512 KB
  bf16_t* Bfrag  = (bf16_t*)(ws + (1024u << 10));       // B*H*L bf16  = 33.5 MB

  // 1) mode parameters
  dss_params_kernel<<<(kH * kN) / 256, 256, 0, stream>>>(log_dt, Lambda, W, params);
  // 2) pack weights to bf16 A-fragments
  pack_w_kernel<<<(kH * kH) / 256, 256, 0, stream>>>(out_weight, Wfrag);
  // 3) scan + skip + GELU -> bf16 B-fragments (8 waves/block, wave per (b,h))
  dss_scan_kernel<<<(kB * kH) / 8, 256, 0, stream>>>(u, params, D, Bfrag);
  // 4) WMMA GEMM + bias (2x2 blocked: 1024 n-pairs x 16 m-pairs of waves)
  dim3 grid(1024 / 8, 16);
  dss_gemm_kernel<<<grid, 256, 0, stream>>>(Wfrag, Bfrag, out_bias, out);
}
